// CorrelationAwareFocalLoss_21131239097026
// MI455X (gfx1250) — matrix operand order repaired and checked by reading, not compiled, not run on hardware
//
#include <hip/hip_runtime.h>
#include <hip/hip_bf16.h>

typedef __attribute__((ext_vector_type(16))) _Float16 v16h;
typedef __attribute__((ext_vector_type(8)))  float    v8f;

#define BATCH        131072
#define CC           64
#define NBLK         128
#define ROWS_PER_BLK (BATCH / NBLK)              // 1024
#define CHUNK_ROWS   64
#define NCHUNK       (ROWS_PER_BLK / CHUNK_ROWS) // 16
#define PITCH_H      72                          // halves per LDS row (144B, 16B-aligned)
// ws layout: [0..4095]=S1 (T^T T), [4096..8191]=M (X^T Y), [8192]=focal sum
#define WS_FLOATS    (2 * 4096 + 1)

union Frag { uint4 u4[2]; v16h h; };

__global__ void zero_ws_kernel(float* __restrict__ ws) {
  int i = blockIdx.x * 256 + threadIdx.x;
  if (i < WS_FLOATS) ws[i] = 0.0f;
}

__global__ __launch_bounds__(256) void main_pass_kernel(
    const float* __restrict__ inp, const float* __restrict__ tgt,
    const float* __restrict__ pw, float* __restrict__ ws) {
  // Transposed tiles: [column][k], padded pitch. Fragment loads become b128.
  __shared__ __align__(16) _Float16 sT[CC * PITCH_H];
  __shared__ __align__(16) _Float16 sX[CC * PITCH_H];
  __shared__ __align__(16) _Float16 sY[CC * PITCH_H];
  __shared__ float red[256];

  const int tid      = threadIdx.x;
  const int lane     = tid & 31;          // wave32
  const int waveId   = tid >> 5;          // 8 waves
  const int mat      = waveId >> 2;       // 0 -> S1 = T^T T, 1 -> M = X^T Y
  const int itile    = waveId & 3;        // output tile-row block
  const int laneHalf = lane >> 4;
  const int lane15   = lane & 15;

  // element-phase mapping: 4 columns x 4 consecutive k rows per thread
  const int c4   = tid & 15;              // float4 column index
  const int col0 = c4 * 4;
  const int kg   = tid >> 4;              // 0..15 -> k rows kg*4 .. kg*4+3
  const float4 pw4 = ((const float4*)pw)[c4];
  const float pws[4] = {pw4.x, pw4.y, pw4.z, pw4.w};

  v8f acc[4] = {};
  float fsum = 0.0f;

  for (int chunk = 0; chunk < NCHUNK; ++chunk) {
    const int base4 = (blockIdx.x * ROWS_PER_BLK + chunk * CHUNK_ROWS) * (CC / 4);
    const float4* inF4 = (const float4*)inp;
    const float4* tgF4 = (const float4*)tgt;

    unsigned int pT[4][2] = {}, pX[4][2] = {}, pY[4][2] = {};
    #pragma unroll
    for (int j = 0; j < 4; ++j) {
      const int k = kg * 4 + j;
      const float4 x4 = inF4[base4 + k * 16 + c4];
      const float4 t4 = tgF4[base4 + k * 16 + c4];
      const float xs[4] = {x4.x, x4.y, x4.z, x4.w};
      const float ts[4] = {t4.x, t4.y, t4.z, t4.w};
      #pragma unroll
      for (int e = 0; e < 4; ++e) {
        const float x = xs[e], t = ts[e];
        const bool tb = (t != 0.0f);
        const bool pb = (x >= 0.0f);          // pred = sigmoid(x) >= 0.5
        // one transcendental pair per element: e = exp(-|x|)
        const float ax  = fabsf(x);
        const float ee  = __expf(-ax);
        const float inv = 1.0f / (1.0f + ee);
        const float prob = pb ? inv : ee * inv;      // sigmoid(x)
        const float l1p  = __logf(1.0f + ee);        // log1p(exp(-|x|)), ee >= ~2e-3
        const float lsp  = fminf(x, 0.0f) - l1p;     // log_sigmoid(x)
        const float lsn  = fminf(-x, 0.0f) - l1p;    // log_sigmoid(-x)
        const float bce  = tb ? (-pws[e] * lsp) : (-lsn);
        const float ompt = tb ? (1.0f - prob) : prob; // 1 - p_t
        fsum += ompt * ompt * bce;
        // exact f16 bit packing of {0,1} values, little-endian half order in k
        const unsigned int th = tb ? 0x3C00u : 0u;
        const unsigned int yh = (tb && pb) ? 0x3C00u : 0u;   // t * pred
        const unsigned int xh = (tb && !pb) ? 0x3C00u : 0u;  // t * (1-pred)
        const int hi = j >> 1, sh = (j & 1) * 16;
        pT[e][hi] |= th << sh; pX[e][hi] |= xh << sh; pY[e][hi] |= yh << sh;
      }
    }
    #pragma unroll
    for (int e = 0; e < 4; ++e) {
      const int off = (col0 + e) * PITCH_H + kg * 4;   // halves; 8B-aligned
      *(uint2*)(sT + off) = make_uint2(pT[e][0], pT[e][1]);
      *(uint2*)(sX + off) = make_uint2(pX[e][0], pX[e][1]);
      *(uint2*)(sY + off) = make_uint2(pY[e][0], pY[e][1]);
    }
    __syncthreads();

    // ---- WMMA over this 64-row K slab; fragments via b128 LDS loads ----
    const _Float16* Asrc = mat ? sX : sT;
    const _Float16* Bsrc = mat ? sY : sT;
    const int mrow = itile * 16 + lane15;   // A-matrix row (original column)
    #pragma unroll
    for (int ks = 0; ks < 2; ++ks) {
      const int kb = ks * 32;
      Frag a;   // A 16x32 f16: VGPR0-3 = K kb+lh*8..+7, VGPR4-7 = +16
      a.u4[0] = *(const uint4*)(Asrc + mrow * PITCH_H + kb + laneHalf * 8);
      a.u4[1] = *(const uint4*)(Asrc + mrow * PITCH_H + kb + 16 + laneHalf * 8);
      #pragma unroll
      for (int n = 0; n < 4; ++n) {
        const int nrow = n * 16 + lane15;
        Frag b; // B 32x16 f16: VGPR0-3 = K kb+lh*8..+7, VGPR4-7 = +16
        b.u4[0] = *(const uint4*)(Bsrc + nrow * PITCH_H + kb + laneHalf * 8);
        b.u4[1] = *(const uint4*)(Bsrc + nrow * PITCH_H + kb + 16 + laneHalf * 8);
        acc[n] = __builtin_amdgcn_wmma_f32_16x16x32_f16(
            false, a.h, false, b.h, (short)0, acc[n], false, false);
      }
    }
    __syncthreads();
  }

  // ---- block reduction of focal sum ----
  red[tid] = fsum;
  __syncthreads();
  for (int s = 128; s > 0; s >>= 1) {
    if (tid < s) red[tid] += red[tid + s];
    __syncthreads();
  }
  if (tid == 0) unsafeAtomicAdd(&ws[2 * 4096], red[0]);

  // ---- accumulate 64x64 Gram matrices (C/D layout: VGPR v -> M=v+8*half) ----
  float* Wmat = ws + mat * 4096;
  #pragma unroll
  for (int n = 0; n < 4; ++n) {
    #pragma unroll
    for (int v = 0; v < 8; ++v) {
      const int row = itile * 16 + v + 8 * laneHalf;
      const int col = n * 16 + lane15;
      unsafeAtomicAdd(&Wmat[row * 64 + col], acc[n][v]);
    }
  }
}

__global__ __launch_bounds__(256) void epilogue_kernel(
    const float* __restrict__ ws, float* __restrict__ out) {
  __shared__ float red[256];
  const int tid = threadIdx.x;
  float s = 0.0f;
  #pragma unroll
  for (int t = 0; t < 16; ++t) {
    const int idx = tid + t * 256;        // 4096 entries of 64x64
    const int i = idx >> 6, j = idx & 63;
    const float corr = ws[idx] * (1.0f / (float)BATCH);
    if (i != j && corr > 0.3f) s += corr * ws[4096 + idx];
  }
  red[tid] = s;
  __syncthreads();
  for (int st = 128; st > 0; st >>= 1) {
    if (tid < st) red[tid] += red[tid + st];
    __syncthreads();
  }
  if (tid == 0)
    out[0] = (ws[2 * 4096] + red[0]) * (1.0f / ((float)BATCH * (float)CC));
}

extern "C" void kernel_launch(void* const* d_in, const int* in_sizes, int n_in,
                              void* d_out, int out_size, void* d_ws, size_t ws_size,
                              hipStream_t stream) {
  const float* inp = (const float*)d_in[0];   // inputs  [B, C] f32
  const float* tgt = (const float*)d_in[1];   // targets [B, C] f32
  const float* pw  = (const float*)d_in[2];   // pos_weights [C] f32
  float* ws  = (float*)d_ws;
  float* out = (float*)d_out;

  zero_ws_kernel<<<(WS_FLOATS + 255) / 256, 256, 0, stream>>>(ws);
  main_pass_kernel<<<NBLK, 256, 0, stream>>>(inp, tgt, pw, ws);
  epilogue_kernel<<<1, 256, 0, stream>>>(ws, out);
}
